// charLM_52261162057797
// MI455X (gfx1250) — matrix operand order repaired and checked by reading, not compiled, not run on hardware
//
#include <hip/hip_runtime.h>
#include <hip/hip_bf16.h>

// ---------------------------------------------------------------- types
typedef _Float16 v16h __attribute__((ext_vector_type(16)));
typedef _Float16 v8h  __attribute__((ext_vector_type(8)));
typedef float    v8f  __attribute__((ext_vector_type(8)));

#define WMMA_F16(a, b, c) \
  __builtin_amdgcn_wmma_f32_16x16x32_f16(false, (a), false, (b), (short)0, (c), false, false)

// ---------------------------------------------------------------- async-to-LDS (CDNA5), guarded
#ifndef __has_builtin
#define __has_builtin(x) 0
#endif
#if __has_builtin(__builtin_amdgcn_global_load_async_to_lds_b128) && \
    __has_builtin(__builtin_amdgcn_s_wait_asynccnt)
#define HAVE_ASYNC 1
typedef int v4i_vec __attribute__((vector_size(16)));
typedef __attribute__((address_space(1))) v4i_vec* as1_v4i;
typedef __attribute__((address_space(3))) v4i_vec* as3_v4i;
#define ASYNC_CP(g, l)                                                    \
  __builtin_amdgcn_global_load_async_to_lds_b128(                         \
      (as1_v4i)(void*)(g), (as3_v4i)(void*)(l), 0, 0)
#define ASYNC_WAIT() __builtin_amdgcn_s_wait_asynccnt(0)
#else
#define HAVE_ASYNC 0
#define ASYNC_CP(g, l) (*(uint4*)(void*)(l) = *(const uint4*)(const void*)(g))
#define ASYNC_WAIT()
#endif

// ---------------------------------------------------------------- fragment loaders
// A fragment (16x32, f16): row = lane&15, halves: K = kb..kb+7 and kb+16..kb+23, kb=(lane>>4)*8
__device__ __forceinline__ v16h load_frag_a(const _Float16* base, int ld, int lane) {
  int row = lane & 15, hi = lane >> 4;
  const _Float16* p = base + (size_t)row * ld + hi * 8;
  union { v16h v; v8h h[2]; } u;
  u.h[0] = *(const v8h*)(p);
  u.h[1] = *(const v8h*)(p + 16);
  return u.v;
}
// B fragment (32x16, f16): col = lane&15, halves: K = kb..kb+7 and kb+16..kb+23, kb=(lane>>4)*8
__device__ __forceinline__ v16h load_frag_b(const _Float16* base, int ld, int lane) {
  int col = lane & 15, hi = lane >> 4;
  const _Float16* p = base + (size_t)col * ld + hi * 8;
  union { v16h v; v8h h[2]; } u;
  u.h[0] = *(const v8h*)(p);
  u.h[1] = *(const v8h*)(p + 16);
  return u.v;
}

// ---------------------------------------------------------------- misc kernels
__global__ void zero_kernel(uint4* p, int n4) {
  int i = blockIdx.x * 256 + threadIdx.x;
  if (i < n4) { uint4 z; z.x = z.y = z.z = z.w = 0u; p[i] = z; }
}

__global__ void cvt_pad_kernel(const float* __restrict__ src, _Float16* __restrict__ dst,
                               int N, int K, int Kpad, int total) {
  int i = blockIdx.x * 256 + threadIdx.x;
  if (i >= total) return;
  int n = i / Kpad, k = i % Kpad;
  float v = (n < N && k < K) ? src[(size_t)n * K + k] : 0.0f;
  dst[i] = (_Float16)v;
}

// ---------------------------------------------------------------- char conv + tanh + max
__global__ __launch_bounds__(128) void conv_kernel(const int* __restrict__ x,
                                                   const float* __restrict__ cemb,
                                                   const float* __restrict__ cw,
                                                   const float* __restrict__ cb,
                                                   float* __restrict__ feat) {
  __shared__ float e[15][24];
  __shared__ int   xw[21];
  int n = blockIdx.x, tid = threadIdx.x;
  if (tid < 21) xw[tid] = x[n * 21 + tid];
  __syncthreads();
  for (int i = tid; i < 315; i += 128) {
    int p = i / 15, c = i % 15;
    e[c][p] = cemb[xw[p] * 15 + c];
  }
  __syncthreads();
  for (int ch = tid; ch < 300; ch += 128) {
    const float* w = cw + (size_t)ch * 45;
    float wr[45];
#pragma unroll
    for (int q = 0; q < 45; ++q) wr[q] = w[q];
    float b = cb[ch];
    float m = -1e30f;
    for (int p = 0; p < 19; ++p) {
      float s = b;
#pragma unroll
      for (int c = 0; c < 15; ++c)
#pragma unroll
        for (int kk = 0; kk < 3; ++kk) s += e[c][p + kk] * wr[c * 3 + kk];
      m = fmaxf(m, tanhf(s));
    }
    feat[(size_t)n * 300 + ch] = m;
  }
}

// ---------------------------------------------------------------- batchnorm (batch stats)
__global__ __launch_bounds__(256) void bn_stats_kernel(const float* __restrict__ feat,
                                                       float* __restrict__ mean,
                                                       float* __restrict__ inv) {
  __shared__ float s1[256], s2[256];
  int ch = blockIdx.x, tid = threadIdx.x;
  float a = 0.f, b = 0.f;
  for (int r = tid; r < 2048; r += 256) {
    float v = feat[(size_t)r * 300 + ch];
    a += v; b += v * v;
  }
  s1[tid] = a; s2[tid] = b;
  __syncthreads();
  for (int st = 128; st > 0; st >>= 1) {
    if (tid < st) { s1[tid] += s1[tid + st]; s2[tid] += s2[tid + st]; }
    __syncthreads();
  }
  if (tid == 0) {
    float mu = s1[0] * (1.0f / 2048.0f);
    float var = s2[0] * (1.0f / 2048.0f) - mu * mu;
    mean[ch] = mu;
    inv[ch]  = rsqrtf(var + 1e-5f);
  }
}

__global__ void bn_apply_kernel(const float* __restrict__ feat, const float* __restrict__ mean,
                                const float* __restrict__ inv, _Float16* __restrict__ out) {
  int i = blockIdx.x * 256 + threadIdx.x;   // 2048*320
  int r = i / 320, c = i % 320;
  float v = 0.f;
  if (c < 300) v = (feat[(size_t)r * 300 + c] - mean[c]) * inv[c];
  out[i] = (_Float16)v;
}

// ---------------------------------------------------------------- fused highway (2 WMMA chains)
__global__ __launch_bounds__(128) void highway_kernel(const _Float16* __restrict__ A16,
                                                      const float* __restrict__ Xin,
                                                      const _Float16* __restrict__ GW,
                                                      const _Float16* __restrict__ HWm,
                                                      const float* __restrict__ gb,
                                                      const float* __restrict__ hb,
                                                      float* __restrict__ outF,
                                                      _Float16* __restrict__ outH) {
  int tid = threadIdx.x, lane = tid & 31;
  int id = blockIdx.x * 4 + (tid >> 5);       // 2432 wave tiles: 128 x 19
  int tm = id / 19, tn = id % 19;
  const _Float16* Ab = A16 + (size_t)tm * 16 * 320;
  const _Float16* Gb = GW  + (size_t)tn * 16 * 320;
  const _Float16* Hb = HWm + (size_t)tn * 16 * 320;
  v8f ag = {}, ah = {};
#pragma unroll
  for (int k = 0; k < 10; ++k) {
    v16h a  = load_frag_a(Ab + k * 32, 320, lane);
    v16h bg = load_frag_b(Gb + k * 32, 320, lane);
    ag = WMMA_F16(a, bg, ag);
    v16h bh = load_frag_b(Hb + k * 32, 320, lane);
    ah = WMMA_F16(a, bh, ah);
  }
  int col  = tn * 16 + (lane & 15);
  int row0 = tm * 16 + (lane >> 4) * 8;
  if (col < 300) {
    float gbv = gb[col], hbv = hb[col];
#pragma unroll
    for (int j = 0; j < 8; ++j) {
      int r = row0 + j;
      float t  = 1.f / (1.f + expf(-(ag[j] + gbv)));
      float hv = fmaxf(ah[j] + hbv, 0.f);
      float xv = Xin[(size_t)r * 300 + col];
      float o  = t * hv + (1.f - t) * xv;
      outF[(size_t)r * 300 + col] = o;
      outH[(size_t)r * 320 + col] = (_Float16)o;
    }
  }
}

// ---------------------------------------------------------------- generic WMMA GEMM (+bias)
// C[M,N] = A[M,Kpad] * B[N,Kpad]^T + bias ; one wave per 16x16 tile, exact grids only.
__global__ __launch_bounds__(128) void gemm_bias_kernel(const _Float16* __restrict__ A, int lda,
                                                        const _Float16* __restrict__ B, int ldb,
                                                        const float* __restrict__ bias,
                                                        float* __restrict__ C, int ldc,
                                                        int tilesN, int ksteps) {
  int tid = threadIdx.x, lane = tid & 31;
  int id = blockIdx.x * 4 + (tid >> 5);
  int tm = id / tilesN, tn = id % tilesN;
  const _Float16* Ab = A + (size_t)tm * 16 * lda;
  const _Float16* Bb = B + (size_t)tn * 16 * ldb;
  v8f acc = {};
  for (int k = 0; k < ksteps; ++k) {
    v16h a = load_frag_a(Ab + k * 32, lda, lane);
    v16h b = load_frag_b(Bb + k * 32, ldb, lane);
    acc = WMMA_F16(a, b, acc);
  }
  int col  = tn * 16 + (lane & 15);
  int row0 = tm * 16 + (lane >> 4) * 8;
  float bv = bias ? bias[col] : 0.f;
#pragma unroll
  for (int j = 0; j < 8; ++j) C[(size_t)(row0 + j) * ldc + col] = acc[j] + bv;
}

// ---------------------------------------------------------------- GRU gate elementwise
__global__ __launch_bounds__(256) void gru_gate_kernel(const float* __restrict__ xproj,
                                                       const float* __restrict__ gh,
                                                       float* __restrict__ hF,
                                                       _Float16* __restrict__ h16,
                                                       _Float16* __restrict__ outs16, int s) {
  int i = blockIdx.x * 256 + threadIdx.x;   // 32*512
  int b = i >> 9, k = i & 511;
  const float* xp = xproj + (size_t)(b * 64 + s) * 1536;
  const float* g  = gh + (size_t)b * 1536;
  float r  = 1.f / (1.f + expf(-(xp[k] + g[k])));
  float z  = 1.f / (1.f + expf(-(xp[512 + k] + g[512 + k])));
  float nn = tanhf(xp[1024 + k] + r * g[1024 + k]);
  float hn = (1.f - z) * nn + z * hF[i];
  hF[i]  = hn;
  h16[i] = (_Float16)hn;
  outs16[(size_t)(b * 64 + s) * 512 + k] = (_Float16)hn;
}

// ---------------------------------------------------------------- flagship logits GEMM
// C[2048,30000] = outs16[2048,512] * linw16[30000,512]^T + lin_b
// BM=64, BN=80 (30000 = 375*80), BK=32; 4 waves, each wave: 1 M-subtile x 5 N-subtiles.
// Double-buffered LDS, async global->LDS when available.
__global__ __launch_bounds__(128) void logits_kernel(const _Float16* __restrict__ A,
                                                     const _Float16* __restrict__ Bw,
                                                     const float* __restrict__ bias,
                                                     float* __restrict__ C) {
  __shared__ __align__(16) _Float16 As[2][64][32];
  __shared__ __align__(16) _Float16 Bs[2][80][32];
  int tid = threadIdx.x, lane = tid & 31, wave = tid >> 5;
  int bid = blockIdx.x;
  int bm = bid / 375, bn = bid % 375;
  const _Float16* Ag = A  + (size_t)bm * 64 * 512;
  const _Float16* Bg = Bw + (size_t)bn * 80 * 512;

  auto issue = [&](int st, int ks) {
#pragma unroll
    for (int i = 0; i < 2; ++i) {           // A tile: 256 x 16B chunks
      int ch = tid + i * 128;
      int row = ch >> 2, ko = (ch & 3) * 8;
      ASYNC_CP(Ag + (size_t)row * 512 + ks * 32 + ko, &As[st][row][ko]);
    }
#pragma unroll
    for (int i = 0; i < 3; ++i) {           // B tile: 320 x 16B chunks
      int ch = tid + i * 128;
      if (ch < 320) {
        int row = ch >> 2, ko = (ch & 3) * 8;
        ASYNC_CP(Bg + (size_t)row * 512 + ks * 32 + ko, &Bs[st][row][ko]);
      }
    }
  };

  v8f acc[5];
#pragma unroll
  for (int n = 0; n < 5; ++n) acc[n] = (v8f){};

  issue(0, 0);
  for (int ks = 0; ks < 16; ++ks) {
    ASYNC_WAIT();
    __syncthreads();
    if (ks + 1 < 16) issue((ks + 1) & 1, ks + 1);
    int st = ks & 1;
    v16h a = load_frag_a(&As[st][wave * 16][0], 32, lane);
#pragma unroll
    for (int n = 0; n < 5; ++n) {
      v16h b = load_frag_b(&Bs[st][n * 16][0], 32, lane);
      acc[n] = WMMA_F16(a, b, acc[n]);
    }
    __syncthreads();
  }

  int col0 = bn * 80;
  int row0 = bm * 64 + wave * 16 + (lane >> 4) * 8;
  int c = lane & 15;
#pragma unroll
  for (int n = 0; n < 5; ++n) {
    int col = col0 + n * 16 + c;
    float bv = bias[col];
#pragma unroll
    for (int j = 0; j < 8; ++j)
      C[(size_t)(row0 + j) * 30000 + col] = acc[n][j] + bv;
  }
}

// ---------------------------------------------------------------- host launcher
extern "C" void kernel_launch(void* const* d_in, const int* in_sizes, int n_in,
                              void* d_out, int out_size, void* d_ws, size_t ws_size,
                              hipStream_t stream) {
  (void)in_sizes; (void)n_in; (void)out_size; (void)ws_size;
  const int*   x     = (const int*)d_in[0];
  const float* cemb  = (const float*)d_in[1];
  const float* convw = (const float*)d_in[2];
  const float* convb = (const float*)d_in[3];
  const float* h1gw  = (const float*)d_in[4];
  const float* h1gb  = (const float*)d_in[5];
  const float* h1hw  = (const float*)d_in[6];
  const float* h1hb  = (const float*)d_in[7];
  const float* h2gw  = (const float*)d_in[8];
  const float* h2gb  = (const float*)d_in[9];
  const float* h2hw  = (const float*)d_in[10];
  const float* h2hb  = (const float*)d_in[11];
  const float* wih   = (const float*)d_in[12];
  const float* whh   = (const float*)d_in[13];
  const float* bih   = (const float*)d_in[14];
  const float* bhh   = (const float*)d_in[15];
  const float* linw  = (const float*)d_in[16];
  const float* linb  = (const float*)d_in[17];

  char* ws = (char*)d_ws;
  size_t cur = 0;
  auto alloc = [&](size_t bytes) -> char* {
    char* p = ws + cur;
    cur += (bytes + 255) & ~(size_t)255;
    return p;
  };

  float*    feat   = (float*)alloc((size_t)2048 * 300 * 4);
  float*    hw1F   = (float*)alloc((size_t)2048 * 300 * 4);
  float*    mean   = (float*)alloc(300 * 4);
  float*    inv    = (float*)alloc(300 * 4);
  _Float16* featH  = (_Float16*)alloc((size_t)2048 * 320 * 2);
  // ---- contiguous zero-init region ----
  char*     zbase  = ws + cur;
  _Float16* hw1H   = (_Float16*)alloc((size_t)2048 * 320 * 2);
  _Float16* embH   = (_Float16*)alloc((size_t)2048 * 320 * 2);
  _Float16* h16    = (_Float16*)alloc((size_t)32 * 512 * 2);
  float*    hF     = (float*)alloc((size_t)32 * 512 * 4);
  size_t    zbytes = (size_t)(ws + cur - zbase);
  // -------------------------------------
  _Float16* wg1    = (_Float16*)alloc((size_t)304 * 320 * 2);
  _Float16* wh1    = (_Float16*)alloc((size_t)304 * 320 * 2);
  _Float16* wg2    = (_Float16*)alloc((size_t)304 * 320 * 2);
  _Float16* wh2    = (_Float16*)alloc((size_t)304 * 320 * 2);
  _Float16* wihH   = (_Float16*)alloc((size_t)1536 * 320 * 2);
  _Float16* whhH   = (_Float16*)alloc((size_t)1536 * 512 * 2);
  _Float16* linH   = (_Float16*)alloc((size_t)30000 * 512 * 2);
  float*    xproj  = (float*)alloc((size_t)2048 * 1536 * 4);
  float*    gh     = (float*)alloc((size_t)32 * 1536 * 4);
  _Float16* outs16 = (_Float16*)alloc((size_t)2048 * 512 * 2);

  float* logits = (float*)d_out;
  float* embF   = (float*)d_out + (size_t)2048 * 30000;

  // 0) zero the state/padded activation region
  int n4 = (int)(zbytes / 16);
  zero_kernel<<<(n4 + 255) / 256, 256, 0, stream>>>((uint4*)zbase, n4);

  // 1) weight conversions to f16 (padded)
  auto cvt = [&](const float* s, _Float16* d, int N, int K, int Kpad) {
    int total = ((N + 15) & ~15) * Kpad;
    cvt_pad_kernel<<<(total + 255) / 256, 256, 0, stream>>>(s, d, N, K, Kpad, total);
  };
  cvt(h1gw, wg1, 300, 300, 320);
  cvt(h1hw, wh1, 300, 300, 320);
  cvt(h2gw, wg2, 300, 300, 320);
  cvt(h2hw, wh2, 300, 300, 320);
  cvt(wih, wihH, 1536, 300, 320);
  cvt(whh, whhH, 1536, 512, 512);
  cvt(linw, linH, 30000, 512, 512);

  // 2) char embedding + conv + tanh + max-pool
  conv_kernel<<<2048, 128, 0, stream>>>(x, cemb, convw, convb, feat);

  // 3) batch-norm statistics + apply (+ f16 pad-to-320 conversion)
  bn_stats_kernel<<<300, 256, 0, stream>>>(feat, mean, inv);
  bn_apply_kernel<<<(2048 * 320) / 256, 256, 0, stream>>>(feat, mean, inv, featH);

  // 4) highway x2 (fused dual-WMMA)
  highway_kernel<<<608, 128, 0, stream>>>(featH, feat, wg1, wh1, h1gb, h1hb, hw1F, hw1H);
  highway_kernel<<<608, 128, 0, stream>>>(hw1H, hw1F, wg2, wh2, h2gb, h2hb, embF, embH);

  // 5) GRU input projection: [2048,320]x[1536,320]^T (128 x 96 tiles)
  gemm_bias_kernel<<<(128 * 96) / 4, 128, 0, stream>>>(embH, 320, wihH, 320, bih,
                                                       xproj, 1536, 96, 10);

  // 6) GRU recurrence: 64 sequential (GEMM + gates) steps
  for (int s = 0; s < 64; ++s) {
    gemm_bias_kernel<<<(2 * 96) / 4, 128, 0, stream>>>(h16, 512, whhH, 512, bhh,
                                                       gh, 1536, 96, 16);
    gru_gate_kernel<<<(32 * 512) / 256, 256, 0, stream>>>(xproj, gh, hF, h16, outs16, s);
  }

  // 7) logits: [2048,512]x[30000,512]^T, LDS double-buffered WMMA
  logits_kernel<<<32 * 375, 128, 0, stream>>>(outs16, linH, linb, logits);
}
